// HierarchicalSchedulabilityModel_33028298506633
// MI455X (gfx1250) — matrix operand order repaired and checked by reading, not compiled, not run on hardware
//
#include <hip/hip_runtime.h>
#include <hip/hip_bf16.h>

#define DEV __device__ __forceinline__

typedef __attribute__((ext_vector_type(16))) _Float16 v16h;
typedef __attribute__((ext_vector_type(8)))  float    v8f;

static constexpr int BSZ   = 64;
static constexpr int SEQL  = 4096;
static constexpr int EMB   = 128;
static constexpr int CHUNK = 32;
static constexpr int HEADS = 4;
static constexpr int NCH   = 128;              // 4096/32
static constexpr float NEGV = -1e9f;
static constexpr float SC_DH  = 0.17677669529663687f;  // 1/sqrt(32)
static constexpr float SC_EMB = 0.08838834764831845f;  // 1/sqrt(128)

union AF { v16h v; _Float16 h[16]; unsigned u[8]; };

// ---- WMMA fragment loaders (wave32; A and B share contiguous k order k=16h+e) ----------
// A 16x32 f16: lane l -> row m0+(l&15); half h=l>>4; u[v]: K=16h+2v (element e -> K=16h+e)
DEV v16h ld_a(const _Float16* A, int lda, int m0, int k0) {
  int l = threadIdx.x & 31; int m = m0 + (l & 15); int hh = l >> 4;
  const unsigned* row = (const unsigned*)(A + (size_t)m * lda + k0);
  AF f;
#pragma unroll
  for (int v = 0; v < 4; ++v) f.u[v] = row[8 * hh + v];
#pragma unroll
  for (int v = 0; v < 4; ++v) f.u[4 + v] = row[8 * hh + 4 + v];
  return f.v;
}
// B 32x16 f16 from row-major (KxN): lane l -> col n0+(l&15); element e -> K = 16h+e
DEV v16h ld_b_rm(const _Float16* Bm, int ldb, int k0, int n0) {
  int l = threadIdx.x & 31; int n = n0 + (l & 15); int hh = l >> 4;
  AF f;
#pragma unroll
  for (int e = 0; e < 16; ++e) f.h[e] = Bm[(size_t)(k0 + 16 * hh + e) * ldb + n];
  return f.v;
}
// B from packed fragment-major buffer: one contiguous 32B load per lane
DEV v16h ld_b_pack(const _Float16* Bp, int Nt, int ks, int nt) {
  int l = threadIdx.x & 31;
  return *(const v16h*)(Bp + (((size_t)(ks * Nt + nt) * 32 + l) << 4));
}
DEV v8f wmma16(v16h a, v16h b, v8f c) {
  return __builtin_amdgcn_wmma_f32_16x16x32_f16(false, a, false, b, (short)0, c, false, false);
}
DEV v8f acc_init(const float* bias, int n0) {
  v8f c;
  float bv = bias ? bias[n0 + (threadIdx.x & 15)] : 0.f;
#pragma unroll
  for (int r = 0; r < 8; ++r) c[r] = bv;
  return c;
}
// Tiled GEMM: A (MxK f16 in LDS), B packed (KxN), per-wave tile jobs.
template <class W>
DEV void gemm_lds(const _Float16* A, int lda, int M, int K,
                  const _Float16* Bp, int N, const float* bias, W wr) {
  int wave = threadIdx.x >> 5, nw = blockDim.x >> 5;
  int Mt = M >> 4, Nt = N >> 4, Ks = K >> 5;
  for (int job = wave; job < Mt * Nt; job += nw) {
    int mt = job / Nt, nt = job - mt * Nt;
    v8f c = acc_init(bias, nt * 16);
    for (int ks = 0; ks < Ks; ++ks)
      c = wmma16(ld_a(A, lda, mt * 16, ks * 32), ld_b_pack(Bp, Nt, ks, nt), c);
    wr(mt * 16, nt * 16, c);
  }
}
// Row LayerNorm over 32 rows x Wd cols (f32 in LDS), 4 threads/row, blockDim=128
DEV void ln_rows(float* X, int Wd, const float* g, const float* bb,
                 float* red, _Float16* o16, int ldo) {
  int t = threadIdx.x; int r = t >> 2, p = t & 3; int seg = Wd >> 2;
  float s = 0.f, ss = 0.f;
  for (int i = p * seg; i < (p + 1) * seg; ++i) { float v = X[r * Wd + i]; s += v; ss += v * v; }
  red[r * 8 + p] = s; red[r * 8 + 4 + p] = ss;
  __syncthreads();
  float S  = red[r * 8] + red[r * 8 + 1] + red[r * 8 + 2] + red[r * 8 + 3];
  float SS = red[r * 8 + 4] + red[r * 8 + 5] + red[r * 8 + 6] + red[r * 8 + 7];
  float inv = 1.f / Wd, mu = S * inv;
  float rs = rsqrtf(SS * inv - mu * mu + 1e-5f);
  for (int i = p * seg; i < (p + 1) * seg; ++i) {
    float v = (X[r * Wd + i] - mu) * rs * g[i] + bb[i];
    X[r * Wd + i] = v;
    if (o16) o16[r * ldo + i] = (_Float16)v;
  }
  __syncthreads();
}
DEV void ln_rows16(_Float16* X, int Wd, const float* g, const float* bb, float* red) {
  int t = threadIdx.x; int r = t >> 2, p = t & 3; int seg = Wd >> 2;
  float s = 0.f, ss = 0.f;
  for (int i = p * seg; i < (p + 1) * seg; ++i) { float v = (float)X[r * Wd + i]; s += v; ss += v * v; }
  red[r * 8 + p] = s; red[r * 8 + 4 + p] = ss;
  __syncthreads();
  float S  = red[r * 8] + red[r * 8 + 1] + red[r * 8 + 2] + red[r * 8 + 3];
  float SS = red[r * 8 + 4] + red[r * 8 + 5] + red[r * 8 + 6] + red[r * 8 + 7];
  float inv = 1.f / Wd, mu = S * inv;
  float rs = rsqrtf(SS * inv - mu * mu + 1e-5f);
  for (int i = p * seg; i < (p + 1) * seg; ++i) {
    float v = ((float)X[r * Wd + i] - mu) * rs * g[i] + bb[i];
    X[r * Wd + i] = (_Float16)v;
  }
  __syncthreads();
}

// ---- K0: pack weight (KxN f32) into WMMA-B fragment-major f16 --------------
__global__ void pack_w16(const float* __restrict__ W, _Float16* __restrict__ P, int K, int N) {
  int tid = blockIdx.x * blockDim.x + threadIdx.x;
  if (tid >= K * N) return;
  int e = tid & 15, lane = (tid >> 4) & 31, rem = tid >> 9;
  int Nt = N >> 4;
  int nt = rem % Nt, ks = rem / Nt;
  int k = ks * 32 + ((lane >> 4) << 4) + e;
  int n = nt * 16 + (lane & 15);
  P[tid] = (_Float16)W[(size_t)k * N + n];
}

// ---- K1: fused per-chunk local path (embed -> local block -> window pool) --
__global__ __launch_bounds__(128) void local_chunk_kernel(
    const float* __restrict__ tasks, const int* __restrict__ lengths,
    const float* __restrict__ pt_w1, const float* __restrict__ pt_b1,
    const float* __restrict__ pt_lng, const float* __restrict__ pt_lnb,
    const _Float16* __restrict__ w2p, const float* __restrict__ pt_b2,
    const _Float16* __restrict__ qkvp, const float* __restrict__ qkv_b,
    const _Float16* __restrict__ outp, const float* __restrict__ out_b,
    const float* __restrict__ n1g, const float* __restrict__ n1b,
    const _Float16* __restrict__ m1p, const float* __restrict__ m1_b,
    const _Float16* __restrict__ m2p, const float* __restrict__ m2_b,
    const float* __restrict__ n2g, const float* __restrict__ n2b,
    const float* __restrict__ relb, const float* __restrict__ wpq,
    const float* __restrict__ pos_emb,
    float* __restrict__ gbuf, _Float16* __restrict__ gh16) {
  __shared__ _Float16 sA[32 * 256];   // staging A (h / q+kT / ln(x))
  __shared__ _Float16 sB[32 * 256];   // staging B (x / probs+attnout / mlp hidden)
  __shared__ _Float16 sV[32 * 128];
  __shared__ float    sX[32 * 128];   // f32 residual stream
  __shared__ float    sRed[288];
  __shared__ float    sRelB[132];

  int blk = blockIdx.x, b = blk >> 7, c = blk & 127;
  int len = lengths[b], base = c * CHUNK;
  int tid = threadIdx.x, lane = tid & 31, hh = lane >> 4, nn = lane & 15, wave = tid >> 5;

  if (tid < 132) sRelB[tid] = relb[tid];

  // 1) h = relu(tasks @ w1 + b1)  (K=2, scalar)
  for (int idx = tid; idx < 32 * 256; idx += 128) {
    int r = idx >> 8, cc = idx & 255;
    const float* tp = tasks + ((size_t)(b * SEQL + base + r)) * 2;
    float v = tp[0] * pt_w1[cc] + tp[1] * pt_w1[256 + cc] + pt_b1[cc];
    sA[idx] = (_Float16)fmaxf(v, 0.f);
  }
  __syncthreads();
  ln_rows16(sA, 256, pt_lng, pt_lnb, sRed);

  // 2) x = h @ w2 + b2  -> sX (f32) + sB (f16)
  gemm_lds(sA, 256, 32, 256, w2p, 128, pt_b2, [&](int m0, int n0, v8f ca) {
    int col = n0 + nn, mb = m0 + 8 * hh;
#pragma unroll
    for (int r = 0; r < 8; ++r) { float v = ca[r]; sX[(mb + r) * 128 + col] = v; sB[(mb + r) * 128 + col] = (_Float16)v; }
  });
  __syncthreads();

  // 3) qkv = x @ qkv_w + b ; q -> sA[0:4096), kT -> sA[4096:), v -> sV
  //    (scalar branch: n0 is tile-aligned so the q/k/v split is wave-uniform)
  gemm_lds(sB, 128, 32, 128, qkvp, 384, qkv_b, [&](int m0, int n0, v8f ca) {
    int nsel = __builtin_amdgcn_readfirstlane(n0);
    int col = n0 + nn, mb = m0 + 8 * hh;
    if (nsel < 128) {
#pragma unroll
      for (int r = 0; r < 8; ++r) sA[(mb + r) * 128 + col] = (_Float16)ca[r];
    } else if (nsel < 256) {
#pragma unroll
      for (int r = 0; r < 8; ++r) sA[4096 + (col - 128) * 32 + (mb + r)] = (_Float16)ca[r];
    } else {
#pragma unroll
      for (int r = 0; r < 8; ++r) sV[(mb + r) * 128 + (col - 256)] = (_Float16)ca[r];
    }
  });
  __syncthreads();

  // 4) scores + bias + mask + softmax (wave == head), probs -> sB[0:4096)
  {
    int head = wave;
    v8f s[2][2];
    v16h b0 = ld_b_rm(sA + 4096 + head * 1024, 32, 0, 0);
    v16h b1 = ld_b_rm(sA + 4096 + head * 1024, 32, 0, 16);
    v16h a0 = ld_a(sA, 128, 0, head * 32);
    v16h a1 = ld_a(sA, 128, 16, head * 32);
    v8f z = {};
    s[0][0] = wmma16(a0, b0, z); s[0][1] = wmma16(a0, b1, z);
    s[1][0] = wmma16(a1, b0, z); s[1][1] = wmma16(a1, b1, z);
#pragma unroll
    for (int mt = 0; mt < 2; ++mt) {
#pragma unroll
      for (int nt = 0; nt < 2; ++nt)
#pragma unroll
        for (int r = 0; r < 8; ++r) {
          int qrow = mt * 16 + 8 * hh + r, key = nt * 16 + nn;
          float v = s[mt][nt][r] * SC_DH;
          int rel = key - qrow; rel = rel < -16 ? -16 : (rel > 16 ? 16 : rel);
          v += sRelB[(rel + 16) * HEADS + head];
          if (base + key >= len) v = NEGV;
          s[mt][nt][r] = v;
        }
      float mx[8], sm[8];
#pragma unroll
      for (int r = 0; r < 8; ++r) mx[r] = fmaxf(s[mt][0][r], s[mt][1][r]);
      for (int off = 1; off < 16; off <<= 1)
#pragma unroll
        for (int r = 0; r < 8; ++r) mx[r] = fmaxf(mx[r], __shfl_xor(mx[r], off, 32));
#pragma unroll
      for (int r = 0; r < 8; ++r) {
        s[mt][0][r] = __expf(s[mt][0][r] - mx[r]);
        s[mt][1][r] = __expf(s[mt][1][r] - mx[r]);
        sm[r] = s[mt][0][r] + s[mt][1][r];
      }
      for (int off = 1; off < 16; off <<= 1)
#pragma unroll
        for (int r = 0; r < 8; ++r) sm[r] += __shfl_xor(sm[r], off, 32);
#pragma unroll
      for (int nt = 0; nt < 2; ++nt)
#pragma unroll
        for (int r = 0; r < 8; ++r) {
          int qrow = mt * 16 + 8 * hh + r, key = nt * 16 + nn;
          sB[head * 1024 + qrow * 32 + key] = (_Float16)(s[mt][nt][r] / sm[r]);
        }
    }
  }
  __syncthreads();

  // 5) attn @ v -> attnout f16 at sB[4096:)
  {
    int head = wave;
#pragma unroll
    for (int mt = 0; mt < 2; ++mt)
#pragma unroll
      for (int nt = 0; nt < 2; ++nt) {
        v8f z = {};
        v8f o = wmma16(ld_a(sB + head * 1024, 32, mt * 16, 0),
                       ld_b_rm(sV, 128, 0, head * 32 + nt * 16), z);
        int mb = mt * 16 + 8 * hh, col = head * 32 + nt * 16 + nn;
#pragma unroll
        for (int r = 0; r < 8; ++r) sB[4096 + (mb + r) * 128 + col] = (_Float16)o[r];
      }
  }
  __syncthreads();

  // 6) out projection (+= into residual)
  gemm_lds(sB + 4096, 128, 32, 128, outp, 128, out_b, [&](int m0, int n0, v8f ca) {
    int col = n0 + nn, mb = m0 + 8 * hh;
#pragma unroll
    for (int r = 0; r < 8; ++r) sX[(mb + r) * 128 + col] += ca[r];
  });
  __syncthreads();
  ln_rows(sX, 128, n1g, n1b, sRed, sA, 128);

  // 7) MLP
  gemm_lds(sA, 128, 32, 128, m1p, 256, m1_b, [&](int m0, int n0, v8f ca) {
    int col = n0 + nn, mb = m0 + 8 * hh;
#pragma unroll
    for (int r = 0; r < 8; ++r) sB[(mb + r) * 256 + col] = (_Float16)fmaxf(ca[r], 0.f);
  });
  __syncthreads();
  gemm_lds(sB, 256, 32, 256, m2p, 128, m2_b, [&](int m0, int n0, v8f ca) {
    int col = n0 + nn, mb = m0 + 8 * hh;
#pragma unroll
    for (int r = 0; r < 8; ++r) sX[(mb + r) * 128 + col] += ca[r];
  });
  __syncthreads();
  ln_rows(sX, 128, n2g, n2b, sRed, nullptr, 0);

  // 8) window pooling  (+ pos_emb) -> gbuf/gh16
  {
    int r = tid >> 2, p = tid & 3;
    float s = 0.f;
    for (int i = p * 32; i < p * 32 + 32; ++i) s += sX[r * 128 + i] * wpq[i];
    sRed[r * 8 + p] = s;
    __syncthreads();
    float* sS = sRed + 256;
    if (p == 0) {
      float v = (sRed[r * 8] + sRed[r * 8 + 1] + sRed[r * 8 + 2] + sRed[r * 8 + 3]) * SC_EMB;
      if (base + r >= len) v = NEGV;
      sS[r] = v;
    }
    __syncthreads();
    if (tid < 32) {
      float v = sS[tid], mx = v;
      for (int off = 1; off < 32; off <<= 1) mx = fmaxf(mx, __shfl_xor(mx, off, 32));
      float e = __expf(v - mx), sm = e;
      for (int off = 1; off < 32; off <<= 1) sm += __shfl_xor(sm, off, 32);
      sS[tid] = e / sm;
    }
    __syncthreads();
    float acc = 0.f;
    for (int r2 = 0; r2 < 32; ++r2) acc += sS[r2] * sX[r2 * 128 + tid];
    acc += pos_emb[c * 128 + tid];
    size_t gi = ((size_t)(b * 128 + c)) * 128 + tid;
    gbuf[gi] = acc; gh16[gi] = (_Float16)acc;
  }
}

// ---- Global layer kernels --------------------------------------------------
__global__ __launch_bounds__(128) void g_qkv_kernel(
    const _Float16* __restrict__ gh16, const _Float16* __restrict__ qkvp,
    const float* __restrict__ qkv_b, _Float16* __restrict__ q16,
    _Float16* __restrict__ kT16, _Float16* __restrict__ v16) {
  __shared__ _Float16 sPA[32 * 128];
  int p = blockIdx.x, tid = threadIdx.x, nn = tid & 15, hh = (tid >> 4) & 1;
  const unsigned* src = (const unsigned*)(gh16 + (size_t)p * 4096);
  unsigned* dst = (unsigned*)sPA;
  for (int i = tid; i < 2048; i += 128) dst[i] = src[i];
  __syncthreads();
  gemm_lds(sPA, 128, 32, 128, qkvp, 384, qkv_b, [&](int m0, int n0, v8f ca) {
    int nsel = __builtin_amdgcn_readfirstlane(n0);
    int col = n0 + nn, mb = m0 + 8 * hh;
    if (nsel < 128) {
#pragma unroll
      for (int r = 0; r < 8; ++r) q16[(size_t)(p * 32 + mb + r) * 128 + col] = (_Float16)ca[r];
    } else if (nsel < 256) {
      int d = col - 128;
#pragma unroll
      for (int r = 0; r < 8; ++r) {
        int R = p * 32 + mb + r, bb = R >> 7, t = R & 127;
        kT16[(((size_t)(bb * 4 + (d >> 5)) * 32) + (d & 31)) * 128 + t] = (_Float16)ca[r];
      }
    } else {
#pragma unroll
      for (int r = 0; r < 8; ++r) v16[(size_t)(p * 32 + mb + r) * 128 + (col - 256)] = (_Float16)ca[r];
    }
  });
}

__global__ __launch_bounds__(128) void g_attn_kernel(
    const _Float16* __restrict__ q16, const _Float16* __restrict__ kT16,
    const _Float16* __restrict__ v16, const float* __restrict__ relb,
    const int* __restrict__ lengths, _Float16* __restrict__ attn16) {
  __shared__ _Float16 sP[4 * 16 * 128];   // probs staging (per wave)
  __shared__ _Float16 sK[32 * 128];       // kT head slice (d x key)
  __shared__ _Float16 sVh[128 * 32];      // v head slice (key x d)
  __shared__ float    sRelB[1028];
  int blk = blockIdx.x, b = blk >> 2, head = blk & 3;
  int len = lengths[b];
  int tid = threadIdx.x, wave = tid >> 5, lane = tid & 31, nn = lane & 15, hh = lane >> 4;
  const _Float16* qb = q16 + (size_t)b * 16384;
  // stage k/v head slices + rel-bias table
  {
    const unsigned* kb = (const unsigned*)(kT16 + (size_t)(b * 4 + head) * 4096);
    unsigned* dk = (unsigned*)sK;
    for (int i = tid; i < 2048; i += 128) dk[i] = kb[i];
    unsigned* dv = (unsigned*)sVh;
    for (int i = tid; i < 2048; i += 128) {
      int t = i >> 4, cp = i & 15;
      dv[i] = *(const unsigned*)(v16 + (size_t)(b * 128 + t) * 128 + head * 32 + cp * 2);
    }
    for (int i = tid; i < 1028; i += 128) sRelB[i] = relb[i];
  }
  __syncthreads();
  for (int jj = 0; jj < 2; ++jj) {
    int m0 = (wave + jj * 4) * 16;
    v8f s[8];
    v16h af = ld_a(qb, 128, m0, head * 32);
#pragma unroll
    for (int nt = 0; nt < 8; ++nt) { v8f z = {}; s[nt] = wmma16(af, ld_b_rm(sK, 128, 0, nt * 16), z); }
#pragma unroll
    for (int nt = 0; nt < 8; ++nt)
#pragma unroll
      for (int r = 0; r < 8; ++r) {
        int qrow = m0 + 8 * hh + r, key = nt * 16 + nn;
        float v = s[nt][r] * SC_DH + sRelB[(key - qrow + 128) * HEADS + head];
        if (key * 32 >= len) v = NEGV;
        s[nt][r] = v;
      }
    float mx[8], sm[8];
#pragma unroll
    for (int r = 0; r < 8; ++r) { mx[r] = s[0][r];
#pragma unroll
      for (int nt = 1; nt < 8; ++nt) mx[r] = fmaxf(mx[r], s[nt][r]); }
    for (int off = 1; off < 16; off <<= 1)
#pragma unroll
      for (int r = 0; r < 8; ++r) mx[r] = fmaxf(mx[r], __shfl_xor(mx[r], off, 32));
#pragma unroll
    for (int r = 0; r < 8; ++r) { sm[r] = 0.f;
#pragma unroll
      for (int nt = 0; nt < 8; ++nt) { s[nt][r] = __expf(s[nt][r] - mx[r]); sm[r] += s[nt][r]; } }
    for (int off = 1; off < 16; off <<= 1)
#pragma unroll
      for (int r = 0; r < 8; ++r) sm[r] += __shfl_xor(sm[r], off, 32);
    _Float16* pp = sP + wave * 2048;
#pragma unroll
    for (int nt = 0; nt < 8; ++nt)
#pragma unroll
      for (int r = 0; r < 8; ++r)
        pp[(8 * hh + r) * 128 + nt * 16 + nn] = (_Float16)(s[nt][r] / sm[r]);
    __syncthreads();
#pragma unroll
    for (int nt2 = 0; nt2 < 2; ++nt2) {
      v8f o = acc_init(nullptr, 0);
#pragma unroll
      for (int ks = 0; ks < 4; ++ks)
        o = wmma16(ld_a(pp, 128, 0, ks * 32), ld_b_rm(sVh, 32, ks * 32, nt2 * 16), o);
      int col = head * 32 + nt2 * 16 + nn;
#pragma unroll
      for (int r = 0; r < 8; ++r)
        attn16[(size_t)(b * 128 + m0 + 8 * hh + r) * 128 + col] = (_Float16)o[r];
    }
    __syncthreads();
  }
}

__global__ __launch_bounds__(128) void g_outln_kernel(
    const _Float16* __restrict__ attn16, const _Float16* __restrict__ outp,
    const float* __restrict__ out_b, const float* __restrict__ n1g,
    const float* __restrict__ n1b, float* __restrict__ gbuf, _Float16* __restrict__ gh16) {
  __shared__ _Float16 sPA[32 * 128];
  __shared__ float sPX[32 * 128];
  __shared__ float sRed[256];
  int p = blockIdx.x, tid = threadIdx.x, nn = tid & 15, hh = (tid >> 4) & 1;
  const unsigned* sa = (const unsigned*)(attn16 + (size_t)p * 4096);
  unsigned* da = (unsigned*)sPA;
  for (int i = tid; i < 2048; i += 128) da[i] = sa[i];
  for (int i = tid; i < 4096; i += 128) sPX[i] = gbuf[(size_t)p * 4096 + i];
  __syncthreads();
  gemm_lds(sPA, 128, 32, 128, outp, 128, out_b, [&](int m0, int n0, v8f ca) {
    int col = n0 + nn, mb = m0 + 8 * hh;
#pragma unroll
    for (int r = 0; r < 8; ++r) sPX[(mb + r) * 128 + col] += ca[r];
  });
  __syncthreads();
  ln_rows(sPX, 128, n1g, n1b, sRed, nullptr, 0);
  for (int i = tid; i < 4096; i += 128) {
    float v = sPX[i]; gbuf[(size_t)p * 4096 + i] = v; gh16[(size_t)p * 4096 + i] = (_Float16)v;
  }
}

__global__ __launch_bounds__(128) void g_mlp_kernel(
    const _Float16* __restrict__ m1p, const float* __restrict__ m1_b,
    const _Float16* __restrict__ m2p, const float* __restrict__ m2_b,
    const float* __restrict__ n2g, const float* __restrict__ n2b,
    float* __restrict__ gbuf, _Float16* __restrict__ gh16) {
  __shared__ _Float16 sPA[32 * 128];
  __shared__ _Float16 sH[32 * 256];
  __shared__ float sPX[32 * 128];
  __shared__ float sRed[256];
  int p = blockIdx.x, tid = threadIdx.x, nn = tid & 15, hh = (tid >> 4) & 1;
  const unsigned* sa = (const unsigned*)(gh16 + (size_t)p * 4096);
  unsigned* da = (unsigned*)sPA;
  for (int i = tid; i < 2048; i += 128) da[i] = sa[i];
  for (int i = tid; i < 4096; i += 128) sPX[i] = gbuf[(size_t)p * 4096 + i];
  __syncthreads();
  gemm_lds(sPA, 128, 32, 128, m1p, 256, m1_b, [&](int m0, int n0, v8f ca) {
    int col = n0 + nn, mb = m0 + 8 * hh;
#pragma unroll
    for (int r = 0; r < 8; ++r) sH[(mb + r) * 256 + col] = (_Float16)fmaxf(ca[r], 0.f);
  });
  __syncthreads();
  gemm_lds(sH, 256, 32, 256, m2p, 128, m2_b, [&](int m0, int n0, v8f ca) {
    int col = n0 + nn, mb = m0 + 8 * hh;
#pragma unroll
    for (int r = 0; r < 8; ++r) sPX[(mb + r) * 128 + col] += ca[r];
  });
  __syncthreads();
  ln_rows(sPX, 128, n2g, n2b, sRed, nullptr, 0);
  for (int i = tid; i < 4096; i += 128) {
    float v = sPX[i]; gbuf[(size_t)p * 4096 + i] = v; gh16[(size_t)p * 4096 + i] = (_Float16)v;
  }
}

// ---- K5: global pooling + classifier (tiny) --------------------------------
__global__ __launch_bounds__(128) void pool_cls_kernel(
    const float* __restrict__ gbuf, const float* __restrict__ gpq,
    const int* __restrict__ lengths, const float* __restrict__ dom,
    const float* __restrict__ w1, const float* __restrict__ b1,
    const float* __restrict__ w2, const float* __restrict__ b2,
    const float* __restrict__ w3, const float* __restrict__ b3,
    const float* __restrict__ w4, const float* __restrict__ b4,
    float* __restrict__ out) {
  __shared__ float sS[128], sW[128], sO[132], sZ1[128], sZ2[64], sZ3[32];
  int b = blockIdx.x, tid = threadIdx.x;
  int len = lengths[b];
  const float* gb = gbuf + (size_t)b * 16384;
  float s = 0.f;
  for (int i = 0; i < 128; ++i) s += gb[tid * 128 + i] * gpq[i];
  if (tid * 32 >= len) s = NEGV;
  sS[tid] = s;
  __syncthreads();
  float mx = -3.4e38f;
  for (int i = 0; i < 128; ++i) mx = fmaxf(mx, sS[i]);
  float e = __expf(sS[tid] - mx), sum = 0.f;
  for (int i = 0; i < 128; ++i) sum += __expf(sS[i] - mx);
  sW[tid] = e / sum;
  __syncthreads();
  float acc = 0.f;
  for (int t = 0; t < 128; ++t) acc += sW[t] * gb[t * 128 + tid];
  sO[tid] = acc;
  if (tid == 0) sO[128] = dom[b];
  __syncthreads();
  float z = b1[tid];
  for (int i = 0; i < 129; ++i) z += sO[i] * w1[i * 128 + tid];
  sZ1[tid] = fmaxf(z, 0.f);
  __syncthreads();
  if (tid < 64) { float z2 = b2[tid]; for (int i = 0; i < 128; ++i) z2 += sZ1[i] * w2[i * 64 + tid]; sZ2[tid] = fmaxf(z2, 0.f); }
  __syncthreads();
  if (tid < 32) { float z3 = b3[tid]; for (int i = 0; i < 64; ++i) z3 += sZ2[i] * w3[i * 32 + tid]; sZ3[tid] = fmaxf(z3, 0.f); }
  __syncthreads();
  if (tid == 0) {
    float lg = b4[0];
    for (int i = 0; i < 32; ++i) lg += sZ3[i] * w4[i];
    out[b] = 1.f / (1.f + __expf(-lg));
    out[64 + b] = lg;
  }
}

// ---- host ------------------------------------------------------------------
extern "C" void kernel_launch(void* const* d_in, const int* in_sizes, int n_in,
                              void* d_out, int out_size, void* d_ws, size_t ws_size,
                              hipStream_t stream) {
  (void)in_sizes; (void)n_in; (void)out_size; (void)ws_size;
  // Inputs flattened in setup_inputs() insertion order (params nested dicts
  // flattened recursively in insertion order, 'global' list in order).
  const float* tasks   = (const float*)d_in[0];
  const int*   lengths = (const int*)d_in[1];
  const float* dom     = (const float*)d_in[2];
  int idx = 3;
  auto nextf = [&]() { return (const float*)d_in[idx++]; };
  const float* pt_w1 = nextf(); const float* pt_b1 = nextf();
  const float* pt_lng = nextf(); const float* pt_lnb = nextf();
  const float* pt_w2 = nextf(); const float* pt_b2 = nextf();
  struct BlkP { const float *qkv_w, *qkv_b, *out_w, *out_b, *n1g, *n1b, *m1w, *m1b, *m2w, *m2b, *n2g, *n2b, *relb; };
  auto rdblk = [&]() {
    BlkP bl;
    bl.qkv_w = nextf(); bl.qkv_b = nextf(); bl.out_w = nextf(); bl.out_b = nextf();
    bl.n1g = nextf(); bl.n1b = nextf(); bl.m1w = nextf(); bl.m1b = nextf();
    bl.m2w = nextf(); bl.m2b = nextf(); bl.n2g = nextf(); bl.n2b = nextf();
    bl.relb = nextf();
    return bl;
  };
  BlkP lb = rdblk();
  BlkP gbl[2] = { rdblk(), rdblk() };
  const float* wpq = nextf(); const float* pos = nextf(); const float* gpq = nextf();
  const float* cw1 = nextf(); const float* cb1 = nextf();
  const float* cw2 = nextf(); const float* cb2 = nextf();
  const float* cw3 = nextf(); const float* cb3 = nextf();
  const float* cw4 = nextf(); const float* cb4 = nextf();

  // workspace carve (~15 MB total)
  char* ws = (char*)d_ws; size_t off = 0;
  auto carve = [&](size_t bytes) -> char* {
    off = (off + 255) & ~(size_t)255; char* p = ws + off; off += bytes; return p;
  };
  _Float16* p_w2   = (_Float16*)carve(32768 * 2);
  _Float16* p_lqkv = (_Float16*)carve(49152 * 2);
  _Float16* p_lout = (_Float16*)carve(16384 * 2);
  _Float16* p_lm1  = (_Float16*)carve(32768 * 2);
  _Float16* p_lm2  = (_Float16*)carve(32768 * 2);
  _Float16 *p_gqkv[2], *p_gout[2], *p_gm1[2], *p_gm2[2];
  for (int l = 0; l < 2; ++l) {
    p_gqkv[l] = (_Float16*)carve(49152 * 2);
    p_gout[l] = (_Float16*)carve(16384 * 2);
    p_gm1[l]  = (_Float16*)carve(32768 * 2);
    p_gm2[l]  = (_Float16*)carve(32768 * 2);
  }
  float*     gbuf   = (float*)carve((size_t)8192 * 128 * 4);
  _Float16*  gh16   = (_Float16*)carve((size_t)8192 * 128 * 2);
  _Float16*  q16    = (_Float16*)carve((size_t)8192 * 128 * 2);
  _Float16*  kT16   = (_Float16*)carve((size_t)64 * 4 * 32 * 128 * 2);
  _Float16*  v16    = (_Float16*)carve((size_t)8192 * 128 * 2);
  _Float16*  attn16 = (_Float16*)carve((size_t)8192 * 128 * 2);

  auto pack = [&](const float* W, _Float16* P, int K, int N) {
    int total = K * N;
    pack_w16<<<(total + 255) / 256, 256, 0, stream>>>(W, P, K, N);
  };
  pack(pt_w2, p_w2, 256, 128);
  pack(lb.qkv_w, p_lqkv, 128, 384); pack(lb.out_w, p_lout, 128, 128);
  pack(lb.m1w, p_lm1, 128, 256);    pack(lb.m2w, p_lm2, 256, 128);
  for (int l = 0; l < 2; ++l) {
    pack(gbl[l].qkv_w, p_gqkv[l], 128, 384); pack(gbl[l].out_w, p_gout[l], 128, 128);
    pack(gbl[l].m1w, p_gm1[l], 128, 256);    pack(gbl[l].m2w, p_gm2[l], 256, 128);
  }

  local_chunk_kernel<<<BSZ * NCH, 128, 0, stream>>>(
      tasks, lengths, pt_w1, pt_b1, pt_lng, pt_lnb, p_w2, pt_b2,
      p_lqkv, lb.qkv_b, p_lout, lb.out_b, lb.n1g, lb.n1b,
      p_lm1, lb.m1b, p_lm2, lb.m2b, lb.n2g, lb.n2b, lb.relb,
      wpq, pos, gbuf, gh16);

  for (int l = 0; l < 2; ++l) {
    g_qkv_kernel<<<256, 128, 0, stream>>>(gh16, p_gqkv[l], gbl[l].qkv_b, q16, kT16, v16);
    g_attn_kernel<<<BSZ * HEADS, 128, 0, stream>>>(q16, kT16, v16, gbl[l].relb, lengths, attn16);
    g_outln_kernel<<<256, 128, 0, stream>>>(attn16, p_gout[l], gbl[l].out_b, gbl[l].n1g, gbl[l].n1b, gbuf, gh16);
    g_mlp_kernel<<<256, 128, 0, stream>>>(p_gm1[l], gbl[l].m1b, p_gm2[l], gbl[l].m2b, gbl[l].n2g, gbl[l].n2b, gbuf, gh16);
  }

  pool_cls_kernel<<<BSZ, 128, 0, stream>>>(gbuf, gpq, lengths, dom,
                                           cw1, cb1, cw2, cb2, cw3, cb3, cw4, cb4,
                                           (float*)d_out);
}
